// AMGSRN_60215441490083
// MI455X (gfx1250) — compile-verified, matrix operand order checked by reading, not run on hardware
//
#include <hip/hip_runtime.h>

typedef _Float16 half_t;
typedef __attribute__((ext_vector_type(16))) _Float16 v16h;
typedef __attribute__((ext_vector_type(8)))  _Float16 v8h;
typedef __attribute__((ext_vector_type(2)))  _Float16 h2_t;
typedef __attribute__((ext_vector_type(8)))  float    v8f;

#define N_PTS   262144
#define TILES   (N_PTS / 16)     // 16384 tiles of 16 points
#define WPB     4                // waves per block (128 threads)
#define NBLOCKS 1024
#define VOX     262144           // 64^3 voxels per grid
#define PACKED_BYTES ((size_t)64 * VOX * sizeof(h2_t))   // 67 MB

__device__ __forceinline__ int clampi(int v, int lo, int hi) {
    return v < lo ? lo : (v > hi ? hi : v);
}

// Load one 16-half A/B fragment for v_wmma_f32_16x16x32_f16 from an LDS row.
// Layout (CDNA5 ISA 7.12.2, 16-bit A 16x32; B 32x16 mirrors it):
//   lanes 0-15 : K = kb+{0..7} (VGPR0-3), kb+{16..23} (VGPR4-7)
//   lanes 16-31: K = kb+{8..15},          kb+{24..31}
// Both halves are 16B-contiguous -> two ds_load_b128 per lane.
__device__ __forceinline__ v16h load_frag(const half_t* row, int kb, int halfId) {
    const v8h lo = *(const v8h*)(row + kb + halfId * 8);
    const v8h hi = *(const v8h*)(row + kb + 16 + halfId * 8);
    return __builtin_shufflevector(lo, hi, 0,1,2,3,4,5,6,7,8,9,10,11,12,13,14,15);
}

// Repack [g][f][z][y][x] f32 -> [g][voxel][f] interleaved f16 pairs.
// One corner gather then carries both features in a single b32 load.
__global__ void __launch_bounds__(256) repack_grids(
    const float* __restrict__ grids, h2_t* __restrict__ outp)
{
    const size_t idx = (size_t)blockIdx.x * blockDim.x + threadIdx.x; // one per 4 voxels
    const size_t i4  = idx * 4;                 // voxel index, multiple of 4
    const size_t g   = i4 >> 18;                // /262144
    const size_t v   = i4 & (VOX - 1);
    const float4 a = *(const float4*)&grids[(g << 19) + v];          // feature 0
    const float4 b = *(const float4*)&grids[(g << 19) + VOX + v];    // feature 1
    v8h pk;
    pk[0] = (half_t)a.x; pk[1] = (half_t)b.x;
    pk[2] = (half_t)a.y; pk[3] = (half_t)b.y;
    pk[4] = (half_t)a.z; pk[5] = (half_t)b.z;
    pk[6] = (half_t)a.w; pk[7] = (half_t)b.w;
    *(v8h*)&outp[i4] = pk;
}

template <bool PACKED>
__global__ void __launch_bounds__(128) amgsrn_fused(
    const float* __restrict__ xin,    // [N,3]
    const float* __restrict__ Min,    // [64,4,4]
    const void*  __restrict__ gsrc,   // packed h2_t[64][VOX] or raw f32 grids
    const float* __restrict__ W0in,   // [128,64]
    const float* __restrict__ b0in,   // [64]
    const float* __restrict__ W1in,   // [64,64]
    const float* __restrict__ b1in,   // [64]
    const float* __restrict__ W2in,   // [64,1]
    const float* __restrict__ b2in,   // [1]
    float* __restrict__ out)          // [N,1]
{
    // ---- block-shared staged parameters ----
    __shared__ __align__(16) half_t Wt0[64 * 128];   // W0^T: [n][k] 16 KB
    __shared__ __align__(16) half_t Wt1[64 * 64];    // W1^T: [n][k]  8 KB
    __shared__ __align__(16) float  Mlds[64 * 16];   // 4 KB
    __shared__ float b0lds[64], b1lds[64], w2lds[64];
    // ---- per-wave tile buffers ----
    __shared__ __align__(16) half_t featsAll[WPB][16 * 128]; // 16 KB
    __shared__ __align__(16) half_t hAll[WPB][16 * 64];      //  8 KB
    __shared__ __align__(16) half_t h2All[WPB][16 * 64];     //  8 KB

    const int tid = threadIdx.x;
    for (int i = tid; i < 128 * 64; i += blockDim.x) {
        const int k = i >> 6, n = i & 63;
        Wt0[n * 128 + k] = (half_t)W0in[i];
    }
    for (int i = tid; i < 64 * 64; i += blockDim.x) {
        const int k = i >> 6, n = i & 63;
        Wt1[n * 64 + k] = (half_t)W1in[i];
    }
    for (int i = tid; i < 64 * 16; i += blockDim.x) Mlds[i] = Min[i];
    if (tid < 64) {
        b0lds[tid] = b0in[tid];
        b1lds[tid] = b1in[tid];
        w2lds[tid] = W2in[tid];
    }
    __syncthreads();

    const float b2v   = b2in[0];
    const int wid     = tid >> 5;
    const int lane    = tid & 31;
    const int halfId  = lane >> 4;   // 0: lanes 0-15, 1: lanes 16-31
    const int l16     = lane & 15;

    half_t* featsW = featsAll[wid];
    half_t* hW     = hAll[wid];
    half_t* h2W    = h2All[wid];

    const int waveGlobal = blockIdx.x * WPB + wid;
    const int waveStride = gridDim.x * WPB;

    for (int tile = waveGlobal; tile < TILES; tile += waveStride) {
        const int pidx = tile * 16 + l16;
        const float px = xin[pidx * 3 + 0];
        const float py = xin[pidx * 3 + 1];
        const float pz = xin[pidx * 3 + 2];

        // ===== stage 1: transform + trilinear gather (L2-resident grids) =====
        // lane (l16, halfId) handles point l16, grids {2t+halfId}, both features
        #pragma unroll 1
        for (int t = 0; t < 32; ++t) {
            const int g = 2 * t + halfId;
            const float* Mg = &Mlds[g * 16];
            const float tx = Mg[0] * px + Mg[1] * py + Mg[2]  * pz + Mg[3];
            const float ty = Mg[4] * px + Mg[5] * py + Mg[6]  * pz + Mg[7];
            const float tz = Mg[8] * px + Mg[9] * py + Mg[10] * pz + Mg[11];
            const float ix = (tx + 1.f) * 0.5f * 63.f;
            const float iy = (ty + 1.f) * 0.5f * 63.f;
            const float iz = (tz + 1.f) * 0.5f * 63.f;
            const float xf = floorf(ix), yf = floorf(iy), zf = floorf(iz);
            const float fx = ix - xf, fy = iy - yf, fz = iz - zf;
            const int x0 = (int)xf, y0 = (int)yf, z0 = (int)zf;

            // per-axis weights with validity folded in (zero weight kills
            // any clamped out-of-range corner, matching the reference mask)
            const float wx0 = (1.f - fx) * ((x0 >= 0 && x0 < 64) ? 1.f : 0.f);
            const float wx1 = fx         * ((x0 >= -1 && x0 < 63) ? 1.f : 0.f);
            const float wy0 = (1.f - fy) * ((y0 >= 0 && y0 < 64) ? 1.f : 0.f);
            const float wy1 = fy         * ((y0 >= -1 && y0 < 63) ? 1.f : 0.f);
            const float wz0 = (1.f - fz) * ((z0 >= 0 && z0 < 64) ? 1.f : 0.f);
            const float wz1 = fz         * ((z0 >= -1 && z0 < 63) ? 1.f : 0.f);
            const int xc0 = clampi(x0, 0, 63),     xc1 = clampi(x0 + 1, 0, 63);
            const int yc0 = clampi(y0, 0, 63),     yc1 = clampi(y0 + 1, 0, 63);
            const int zc0 = clampi(z0, 0, 63),     zc1 = clampi(z0 + 1, 0, 63);

            const float wzy00 = wz0 * wy0, wzy01 = wz0 * wy1;
            const float wzy10 = wz1 * wy0, wzy11 = wz1 * wy1;
            const int rb00 = (zc0 << 12) + (yc0 << 6);
            const int rb01 = (zc0 << 12) + (yc1 << 6);
            const int rb10 = (zc1 << 12) + (yc0 << 6);
            const int rb11 = (zc1 << 12) + (yc1 << 6);

            const h2_t*  gbp = PACKED ? ((const h2_t*)gsrc) + (size_t)g * VOX
                                      : (const h2_t*)nullptr;
            const float* gbf = PACKED ? (const float*)nullptr
                                      : ((const float*)gsrc) + (size_t)g * 2u * VOX;

            float f0 = 0.f, f1 = 0.f;
            auto corner = [&](int off, float w) {
                if (PACKED) {
                    const h2_t v = gbp[off];          // one b32 gather, both feats
                    f0 = fmaf(w, (float)v[0], f0);
                    f1 = fmaf(w, (float)v[1], f1);
                } else {
                    f0 = fmaf(w, gbf[off], f0);
                    f1 = fmaf(w, gbf[off + VOX], f1);
                }
            };
            corner(rb00 + xc0, wzy00 * wx0);
            corner(rb00 + xc1, wzy00 * wx1);
            corner(rb01 + xc0, wzy01 * wx0);
            corner(rb01 + xc1, wzy01 * wx1);
            corner(rb10 + xc0, wzy10 * wx0);
            corner(rb10 + xc1, wzy10 * wx1);
            corner(rb11 + xc0, wzy11 * wx0);
            corner(rb11 + xc1, wzy11 * wx1);

            featsW[l16 * 128 + g * 2 + 0] = (half_t)f0;
            featsW[l16 * 128 + g * 2 + 1] = (half_t)f1;
        }
        __builtin_amdgcn_wave_barrier();

        // ===== layer 0: [16x128] x [128x64], WMMA f16->f32, bias+ReLU =====
        v16h a0[4];
        #pragma unroll
        for (int kc = 0; kc < 4; ++kc)
            a0[kc] = load_frag(featsW + l16 * 128, kc * 32, halfId);
        #pragma unroll
        for (int nt = 0; nt < 4; ++nt) {
            v8f c = {};
            #pragma unroll
            for (int kc = 0; kc < 4; ++kc) {
                const v16h b = load_frag(Wt0 + (nt * 16 + l16) * 128, kc * 32, halfId);
                c = __builtin_amdgcn_wmma_f32_16x16x32_f16(
                        false, a0[kc], false, b, (short)0, c, false, false);
            }
            const int n = nt * 16 + l16;
            const float bias = b0lds[n];
            #pragma unroll
            for (int r = 0; r < 8; ++r) {
                const int m = halfId * 8 + r;           // D: VGPR r -> M = 8*halfId + r
                hW[m * 64 + n] = (half_t)fmaxf(c[r] + bias, 0.f);
            }
        }
        __builtin_amdgcn_wave_barrier();

        // ===== layer 1: [16x64] x [64x64], WMMA, bias+ReLU =====
        v16h a1[2];
        #pragma unroll
        for (int kc = 0; kc < 2; ++kc)
            a1[kc] = load_frag(hW + l16 * 64, kc * 32, halfId);
        #pragma unroll
        for (int nt = 0; nt < 4; ++nt) {
            v8f c = {};
            #pragma unroll
            for (int kc = 0; kc < 2; ++kc) {
                const v16h b = load_frag(Wt1 + (nt * 16 + l16) * 64, kc * 32, halfId);
                c = __builtin_amdgcn_wmma_f32_16x16x32_f16(
                        false, a1[kc], false, b, (short)0, c, false, false);
            }
            const int n = nt * 16 + l16;
            const float bias = b1lds[n];
            #pragma unroll
            for (int r = 0; r < 8; ++r) {
                const int m = halfId * 8 + r;
                h2W[m * 64 + n] = (half_t)fmaxf(c[r] + bias, 0.f);
            }
        }
        __builtin_amdgcn_wave_barrier();

        // ===== layer 2: [16x64] x [64x1] — split dot + wave32 shuffle =====
        float acc = 0.f;
        const half_t* hrow = h2W + l16 * 64;
        #pragma unroll
        for (int n = 0; n < 32; ++n) {
            const int nn = halfId * 32 + n;
            acc += (float)hrow[nn] * w2lds[nn];
        }
        acc += __shfl_xor(acc, 16);
        if (halfId == 0) out[pidx] = acc + b2v;
        __builtin_amdgcn_wave_barrier();
    }
}

extern "C" void kernel_launch(void* const* d_in, const int* in_sizes, int n_in,
                              void* d_out, int out_size, void* d_ws, size_t ws_size,
                              hipStream_t stream) {
    const float* x     = (const float*)d_in[0];
    const float* M     = (const float*)d_in[1];
    const float* grids = (const float*)d_in[2];
    const float* W0    = (const float*)d_in[3];
    const float* b0    = (const float*)d_in[4];
    const float* W1    = (const float*)d_in[5];
    const float* b1    = (const float*)d_in[6];
    const float* W2    = (const float*)d_in[7];
    const float* b2    = (const float*)d_in[8];
    float* out = (float*)d_out;
    (void)in_sizes; (void)n_in; (void)out_size;

    if (ws_size >= PACKED_BYTES) {
        // Repack grids into feature-interleaved f16 (halves L2 gather requests).
        h2_t* packed = (h2_t*)d_ws;
        const int nvox4 = 64 * VOX / 4;               // threads, 4 voxels each
        repack_grids<<<nvox4 / 256, 256, 0, stream>>>(grids, packed);
        amgsrn_fused<true><<<NBLOCKS, WPB * 32, 0, stream>>>(
            x, M, (const void*)packed, W0, b0, W1, b1, W2, b2, out);
    } else {
        amgsrn_fused<false><<<NBLOCKS, WPB * 32, 0, stream>>>(
            x, M, (const void*)grids, W0, b0, W1, b1, W2, b2, out);
    }
}
